// FactorizedEMAQuantizer_81990925680815
// MI455X (gfx1250) — matrix operand order repaired and checked by reading, not compiled, not run on hardware
//
#include <hip/hip_runtime.h>
#include <hip/hip_bf16.h>

// ---------------- problem constants (from reference setup_inputs) -----------
static constexpr int Bb = 16, Cc = 256, Ss = 4096;   // S = H*W*D = 16*16*16
static constexpr int Nn = Bb * Ss;                   // 65536 rows
static constexpr int KS = 1024, KY = 512;
#define DECAY_F 0.99f
#define OMD_F   0.01f
#define EPS_F   1e-5f

// ---------------- WMMA vector types -----------------------------------------
typedef __attribute__((ext_vector_type(16))) __bf16 v16bf;
typedef __attribute__((ext_vector_type(8)))  float  v8f;
union BFVec { v16bf v; uint4 q[2]; };

__device__ __forceinline__ unsigned short f2bf(float x) {
  unsigned int u = __float_as_uint(x);
  unsigned int lsb = (u >> 16) & 1u;
  u += 0x7fffu + lsb;                // round-to-nearest-even
  return (unsigned short)(u >> 16);
}

// ---------------- workspace layout (bytes) ----------------------------------
static constexpr size_t OFF_ZF   = 0;                         // bf16 [N,C]
static constexpr size_t OFF_EMBS = 33554432;                  // bf16 [KS,C]
static constexpr size_t OFF_EMBY = 34078720;                  // bf16 [KY,C]
static constexpr size_t OFF_ENS  = 34340864;                  // f32 [KS]
static constexpr size_t OFF_ENY  = 34344960;                  // f32 [KY]
static constexpr size_t OFF_SIDX = 34347008;                  // i32 [N]
static constexpr size_t OFF_YIDX = 34609152;                  // i32 [N]
static constexpr size_t OFF_BALS = 34871296;                  // f32 [KS]
static constexpr size_t OFF_BALY = 34875392;                  // f32 [KY]
// ---- zeroed-every-call region starts here ----
static constexpr size_t OFF_DWS  = 34877440;                  // f32 [KS,C] (becomes new_emb_s)
static constexpr size_t OFF_DWY  = 35926016;                  // f32 [KY,C] (becomes new_emb_y)
static constexpr size_t OFF_CNTS = 36450304;                  // f32 [KS]
static constexpr size_t OFF_CNTY = 36454400;                  // f32 [KY]
static constexpr size_t OFF_PART = 36456448;                  // f32 [4096]
static constexpr size_t ZERO_BYTES = 36472832 - OFF_DWS;      // 1,595,392 B
static constexpr int    ZERO_U4    = (int)(ZERO_BYTES / 16);  // 99,712 uint4

// ---------------- kernels ---------------------------------------------------
__global__ void k_zero(uint4* p, int n) {
  int g = blockIdx.x * 256 + threadIdx.x;
  if (g < n) p[g] = uint4{0u, 0u, 0u, 0u};
}

// z [b,c,h,w,d] f32  ->  z_flat [n=(b,s), c] bf16   (64x64 LDS transpose tile)
__global__ __launch_bounds__(256) void k_prep_z(const float* __restrict__ z,
                                                unsigned short* __restrict__ zf) {
  __shared__ float tile[64][65];
  int bid = blockIdx.x;              // 16 * 4 * 64 = 4096 blocks
  int b  = bid >> 8;
  int ct = (bid >> 6) & 3;           // c-tile (64 channels)
  int st = bid & 63;                 // s-tile (64 spatial)
  int c0 = ct << 6, s0 = st << 6;
  int t = threadIdx.x;
  int tr = t >> 6, tc = t & 63;
#pragma unroll
  for (int i = 0; i < 16; ++i) {
    int r = (i << 2) + tr;           // channel row within tile
    tile[r][tc] = z[((size_t)(b * Cc + c0 + r) << 12) + s0 + tc];
  }
  __syncthreads();
#pragma unroll
  for (int i = 0; i < 16; ++i) {
    int sr = (i << 2) + tr;          // spatial row within tile
    zf[((size_t)((b << 12) + s0 + sr) << 8) + c0 + tc] = f2bf(tile[tc][sr]);
  }
}

// embeddings f32 -> bf16 + row squared norms
__global__ __launch_bounds__(256) void k_prep_emb(const float* __restrict__ es,
                                                  const float* __restrict__ ey,
                                                  unsigned short* __restrict__ ebs,
                                                  unsigned short* __restrict__ eby,
                                                  float* __restrict__ ens,
                                                  float* __restrict__ eny) {
  int row = blockIdx.x;              // 1536 blocks
  int c = threadIdx.x;
  const float* src; unsigned short* dst; float* en; int r;
  if (row < KS) { src = es; dst = ebs; en = ens; r = row; }
  else          { src = ey; dst = eby; en = eny; r = row - KS; }
  float v = src[r * Cc + c];
  dst[r * Cc + c] = f2bf(v);
  __shared__ float red[256];
  red[c] = v * v;
  __syncthreads();
  for (int s = 128; s > 0; s >>= 1) { if (c < s) red[c] += red[c + s]; __syncthreads(); }
  if (c == 0) en[r] = red[0];
}

// Fused GEMM + argmin for BOTH codebooks.
// Block = 8 waves; wave owns 16 rows. B tiles staged in LDS via async copies,
// double-buffered; A strip lives in registers across both codebooks.
__global__ __launch_bounds__(256) void k_argmin2(
    const unsigned short* __restrict__ zf,
    const unsigned short* __restrict__ embS, const float* __restrict__ enS,
    const unsigned short* __restrict__ embY, const float* __restrict__ enY,
    int* __restrict__ sidx, float* __restrict__ sidxf, float* __restrict__ cntS,
    int* __restrict__ yidx, float* __restrict__ yidxf, float* __restrict__ cntY) {
  __shared__ unsigned short btile[2][16 * 256];   // 2 x 8KB B tiles
  const int tid  = threadIdx.x;
  const int lane = tid & 31;
  const int wid  = tid >> 5;
  const int row_base = blockIdx.x * 128 + wid * 16;
  const int lg   = lane & 15;
  const int kqA  = (lane < 16) ? 0 : 1;   // A layout: hi half-wave holds K+8
  const int kqB  = (lane < 16) ? 0 : 1;   // B layout: hi half-wave holds K+8
  // copy assignment: thread copies 32B of LDS row (col = tid>>4, chunk = tid&15)
  const int cpCol = tid >> 4, cpChunk = tid & 15;
  const unsigned ldsOff = (unsigned)(uintptr_t)(&btile[0][0])
                        + (unsigned)((cpCol << 9) + (cpChunk << 5));

  // Preload A strip: 16 rows x 256 bf16 in ISA 16-bit A layout (once, reused).
  const uint4* arow = reinterpret_cast<const uint4*>(zf + ((size_t)(row_base + lg) << 8));
  BFVec a[8];
#pragma unroll
  for (int i = 0; i < 8; ++i) {
    a[i].q[0] = arow[4 * i + kqA];
    a[i].q[1] = arow[4 * i + 2 + kqA];
  }

  auto run = [&](const unsigned short* __restrict__ emb,
                 const float* __restrict__ enorm, int K,
                 int* idx_out, float* idxf_out, float* cnt) {
    const int T = K >> 4;
    auto issue = [&](int ct, int buf) {   // async-copy one 16x256 bf16 tile
      unsigned long long g = (unsigned long long)(uintptr_t)
          (emb + ((size_t)((ct << 4) + cpCol) << 8) + (cpChunk << 4));
      unsigned l = ldsOff + (unsigned)(buf << 13);
      asm volatile("global_load_async_to_lds_b128 %0, %1, off"
                   :: "v"(l), "v"(g) : "memory");
      asm volatile("global_load_async_to_lds_b128 %0, %1, off offset:16"
                   :: "v"(l), "v"(g) : "memory");
    };
    float minv[8]; int mini[8];
#pragma unroll
    for (int v = 0; v < 8; ++v) { minv[v] = 3.4e38f; mini[v] = 0; }

    issue(0, 0);
    for (int ct = 0; ct < T; ++ct) {
      asm volatile("s_wait_asynccnt 0x0" ::: "memory");  // my tile-copy done
      __syncthreads();                                    // all waves' copies done
      if (ct + 1 < T) issue(ct + 1, (ct + 1) & 1);        // prefetch next tile
      // gather all 8 B fragments first so ds_loads clause-issue,
      // then run the 8 accumulate-chained WMMAs back-to-back
      const uint4* bp = reinterpret_cast<const uint4*>(&btile[ct & 1][lg << 8]);
      BFVec bv[8];
#pragma unroll
      for (int i = 0; i < 8; ++i) {
        bv[i].q[0] = bp[4 * i + kqB];
        bv[i].q[1] = bp[4 * i + 2 + kqB];
      }
      v8f acc = {0.f, 0.f, 0.f, 0.f, 0.f, 0.f, 0.f, 0.f};
#pragma unroll
      for (int i = 0; i < 8; ++i)
        acc = __builtin_amdgcn_wmma_f32_16x16x32_bf16(false, a[i].v, false, bv[i].v,
                                                      (short)0, acc, false, false);
      int col = (ct << 4) + lg;
      float en = enorm[col];
#pragma unroll
      for (int v = 0; v < 8; ++v) {       // score = |e|^2 - 2 z.e  (|z|^2 dropped)
        float dval = en - 2.0f * acc[v];
        if (dval < minv[v]) { minv[v] = dval; mini[v] = col; }
      }
    }
    // reduce across the 16 columns held by each 16-lane group (first-index ties)
#pragma unroll
    for (int d = 1; d < 16; d <<= 1) {
#pragma unroll
      for (int v = 0; v < 8; ++v) {
        float ov = __shfl_xor(minv[v], d, 32);
        int   oi = __shfl_xor(mini[v], d, 32);
        if (ov < minv[v] || (ov == minv[v] && oi < mini[v])) { minv[v] = ov; mini[v] = oi; }
      }
    }
#pragma unroll
    for (int v = 0; v < 8; ++v) {
      if (lg == v) {
        int row = row_base + ((lane < 16) ? v : v + 8);
        idx_out[row]  = mini[v];
        idxf_out[row] = (float)mini[v];
        atomicAdd(&cnt[mini[v]], 1.0f);   // fused counts
      }
    }
  };

  run(embS, enS, KS, sidx, sidxf, cntS);
  __syncthreads();
  run(embY, enY, KY, yidx, yidxf, cntY);
}

// dw[k,c] += z_flat[n,c]; coalesced z reads, atomic targets stay in L2.
__global__ __launch_bounds__(256) void k_scatter(const float* __restrict__ z,
                                                 const int* __restrict__ si,
                                                 const int* __restrict__ yi,
                                                 float* __restrict__ dws,
                                                 float* __restrict__ dwy) {
  int bid = blockIdx.x;              // 16 * 256 * 16 = 65536 blocks
  int b  = bid >> 12;
  int c  = (bid >> 4) & 255;
  int sc = bid & 15;
  int s  = (sc << 8) + threadIdx.x;
  int n  = (b << 12) + s;
  float val = z[((size_t)(b * Cc + c) << 12) + s];
  atomicAdd(&dws[si[n] * Cc + c], val);
  atomicAdd(&dwy[yi[n] * Cc + c], val);
}

// per-codebook: new_cs -> nsum -> bal (one block, blockDim == K)
__global__ void k_bal(const float* __restrict__ cnt, const float* __restrict__ cs_in,
                      float* __restrict__ bal, int K) {
  __shared__ float red[1024];
  int k = threadIdx.x;
  float ncs = cs_in[k] * DECAY_F + OMD_F * cnt[k];
  red[k] = ncs;
  __syncthreads();
  for (int s = K >> 1; s > 0; s >>= 1) { if (k < s) red[k] += red[k + s]; __syncthreads(); }
  float nsum = red[0];
  bal[k] = (ncs + EPS_F) / (nsum + (float)K * EPS_F) * nsum;
}

// new_emb = (avg*0.99 + 0.01*dw) / bal   (in place over dw)
__global__ void k_newemb(const float* __restrict__ avgs, const float* __restrict__ avgy,
                         const float* __restrict__ bals, const float* __restrict__ baly,
                         float* __restrict__ dws, float* __restrict__ dwy) {
  int g = blockIdx.x * 256 + threadIdx.x;   // 1536 * 256 threads
  if (g < KS * Cc) {
    float v = avgs[g] * DECAY_F + OMD_F * dws[g];
    dws[g] = v / bals[g >> 8];
  } else {
    int g2 = g - KS * Cc;
    float v = avgy[g2] * DECAY_F + OMD_F * dwy[g2];
    dwy[g2] = v / baly[g2 >> 8];
  }
}

// per (b,h,w) block, thread = channel: z_q + dis partials
__global__ __launch_bounds__(256) void k_out(const int* __restrict__ si,
                                             const int* __restrict__ yi,
                                             const float* __restrict__ nes,
                                             const float* __restrict__ ney,
                                             float* __restrict__ zq,
                                             float* __restrict__ partials) {
  __shared__ int lsi[16], lyi[16];
  __shared__ float red[256];
  int bid = blockIdx.x;              // 4096 blocks
  int b = bid >> 8, hw = bid & 255;
  int n0 = (b << 12) + (hw << 4);
  int t = threadIdx.x;
  if (t < 16) lsi[t] = si[n0 + t];
  else if (t < 32) lyi[t - 16] = yi[n0 + t - 16];
  __syncthreads();
  int c = t;
  float sv[16], yv[16];
  float ss = 0.f, yy = 0.f, sy = 0.f;
#pragma unroll
  for (int d = 0; d < 16; ++d) {
    float s = nes[lsi[d] * Cc + c];
    float y = ney[lyi[d] * Cc + c];
    sv[d] = s; yv[d] = y;
    ss += s * s; yy += y * y; sy += s * y;
  }
  float ns = fmaxf(sqrtf(ss), 1e-12f);
  float ny = fmaxf(sqrtf(yy), 1e-12f);
  float dot = sy / (ns * ny);
  float* o = zq + ((size_t)(b * Cc + c) << 12) + (hw << 4);
#pragma unroll
  for (int d = 0; d < 16; ++d) o[d] = 0.5f * (sv[d] + yv[d]);
  red[t] = dot * dot;
  __syncthreads();
  for (int s2 = 128; s2 > 0; s2 >>= 1) { if (t < s2) red[t] += red[t + s2]; __syncthreads(); }
  if (t == 0) partials[bid] = red[0];
}

__global__ void k_dis(const float* __restrict__ partials, float* __restrict__ out) {
  __shared__ float red[256];
  int t = threadIdx.x;
  float acc = 0.f;
  for (int i = t; i < 4096; i += 256) acc += partials[i];
  red[t] = acc;
  __syncthreads();
  for (int s = 128; s > 0; s >>= 1) { if (t < s) red[t] += red[t + s]; __syncthreads(); }
  if (t == 0) out[0] = red[0] * (1.0f / 1048576.0f);
}

// ---------------- launch ----------------------------------------------------
extern "C" void kernel_launch(void* const* d_in, const int* in_sizes, int n_in,
                              void* d_out, int out_size, void* d_ws, size_t ws_size,
                              hipStream_t stream) {
  const float* z    = (const float*)d_in[0];
  const float* es   = (const float*)d_in[1];
  const float* ey   = (const float*)d_in[2];
  const float* css  = (const float*)d_in[3];
  const float* csy  = (const float*)d_in[4];
  const float* avgs = (const float*)d_in[5];
  const float* avgy = (const float*)d_in[6];

  char* ws = (char*)d_ws;
  unsigned short* zf   = (unsigned short*)(ws + OFF_ZF);
  unsigned short* ebs  = (unsigned short*)(ws + OFF_EMBS);
  unsigned short* eby  = (unsigned short*)(ws + OFF_EMBY);
  float* ens  = (float*)(ws + OFF_ENS);
  float* eny  = (float*)(ws + OFF_ENY);
  int*   sidx = (int*)(ws + OFF_SIDX);
  int*   yidx = (int*)(ws + OFF_YIDX);
  float* bals = (float*)(ws + OFF_BALS);
  float* baly = (float*)(ws + OFF_BALY);
  float* dws  = (float*)(ws + OFF_DWS);
  float* dwy  = (float*)(ws + OFF_DWY);
  float* cnts = (float*)(ws + OFF_CNTS);
  float* cnty = (float*)(ws + OFF_CNTY);
  float* part = (float*)(ws + OFF_PART);

  float* out    = (float*)d_out;
  float* out_zq = out;                         // 16,777,216
  float* out_ds = out + 16777216;              // 1
  float* out_si = out + 16777217;              // 65,536
  float* out_yi = out + 16777217 + 65536;      // 65,536

  k_zero<<<(ZERO_U4 + 255) / 256, 256, 0, stream>>>((uint4*)(ws + OFF_DWS), ZERO_U4);
  k_prep_z<<<4096, 256, 0, stream>>>(z, zf);
  k_prep_emb<<<KS + KY, 256, 0, stream>>>(es, ey, ebs, eby, ens, eny);

  k_argmin2<<<Nn / 128, 256, 0, stream>>>(zf, ebs, ens, eby, eny,
                                          sidx, out_si, cnts,
                                          yidx, out_yi, cnty);

  k_scatter<<<65536, 256, 0, stream>>>(z, sidx, yidx, dws, dwy);

  k_bal<<<1, KS, 0, stream>>>(cnts, css, bals, KS);
  k_bal<<<1, KY, 0, stream>>>(cnty, csy, baly, KY);
  k_newemb<<<(KS + KY), 256, 0, stream>>>(avgs, avgy, bals, baly, dws, dwy);

  k_out<<<4096, 256, 0, stream>>>(sidx, yidx, dws, dwy, out_zq, part);
  k_dis<<<1, 256, 0, stream>>>(part, out_ds);
}
